// Attention_58437325029549
// MI455X (gfx1250) — matrix operand order repaired and checked by reading, not compiled, not run on hardware
//
#include <hip/hip_runtime.h>
#include <hip/hip_bf16.h>

// ---------------------------------------------------------------------------
// GQA attention block for MI455X (gfx1250, wave32, WMMA f16->f32).
// Pipeline: f32->f16 convert, TN WMMA GEMMs (QKV, 32x64/wave), RoPE,
// V transpose, flash-style WMMA attention (2 GQA heads per wave sharing
// K/V fragments), TN WMMA GEMM (output proj, f32 out).
// ---------------------------------------------------------------------------

typedef __attribute__((ext_vector_type(16))) _Float16 v16h;
typedef __attribute__((ext_vector_type(8)))  _Float16 v8h;
typedef __attribute__((ext_vector_type(8)))  float    v8f;

union F16Frag { v16h v; v8h h[2]; };

#define BB   2
#define SS   2048
#define DDim 2048
#define HH   32
#define KVH  8
#define HDD  64
#define BSs  (BB * SS)        // 4096 rows

static __device__ __forceinline__ v8f wmma_f16(const F16Frag& a, const F16Frag& b, v8f c) {
  // D = A(16x32 f16) * B(32x16 f16) + C(16x16 f32)
  return __builtin_amdgcn_wmma_f32_16x16x32_f16(false, a.v, false, b.v, (short)0, c, false, false);
}

// ------------------------------ f32 -> f16 ---------------------------------
__global__ void __launch_bounds__(256)
cvt_f32_f16(const float* __restrict__ in, _Float16* __restrict__ out, int n) {
  int i = blockIdx.x * 256 + threadIdx.x;
  if (i < n) out[i] = (_Float16)in[i];
}

// --------------------- TN GEMM: C(MxN) = A(MxK) * W(NxK)^T -----------------
// One wave computes a 32x64 strip (two 16-row M-tiles sharing B-fragments).
// A-fragment (16x32 f16): lane<16: row=lane, K=k0..k0+15;
// lane>=16: same row set, K=k0+16..k0+31.  B-fragment (32x16): lane
// holds column N with 16 contiguous K values == a W row slice.
template <typename OutT>
__global__ void __launch_bounds__(256)
gemm_tn_wmma(const _Float16* __restrict__ A, const _Float16* __restrict__ W,
             OutT* __restrict__ C, int M, int N, int K) {
  const int lane = threadIdx.x & 31;
  const int wid  = (blockIdx.x * blockDim.x + threadIdx.x) >> 5;
  const int ntiles = N >> 6;                 // 64-wide strips
  const int mt = wid / ntiles;               // 32-row strips
  const int nt = wid - mt * ntiles;
  if (mt * 32 >= M) return;
  const int hi = lane >> 4;
  const int ln = lane & 15;

  const _Float16* arow0 = A + (size_t)(mt * 32 + ln) * K + hi * 16;
  const _Float16* arow1 = arow0 + (size_t)16 * K;
  const _Float16* wrow[4];
#pragma unroll
  for (int t = 0; t < 4; ++t)
    wrow[t] = W + (size_t)(nt * 64 + t * 16 + ln) * K + hi * 16;

  v8f acc[2][4] = {{v8f{}, v8f{}, v8f{}, v8f{}}, {v8f{}, v8f{}, v8f{}, v8f{}}};
  for (int k0 = 0; k0 < K; k0 += 32) {
    F16Frag a0, a1;
    a0.h[0] = *(const v8h*)(arow0 + k0);
    a0.h[1] = *(const v8h*)(arow0 + k0 + 8);
    a1.h[0] = *(const v8h*)(arow1 + k0);
    a1.h[1] = *(const v8h*)(arow1 + k0 + 8);
#pragma unroll
    for (int t = 0; t < 4; ++t) {
      F16Frag bf;
      bf.h[0] = *(const v8h*)(wrow[t] + k0);
      bf.h[1] = *(const v8h*)(wrow[t] + k0 + 8);
      acc[0][t] = wmma_f16(a0, bf, acc[0][t]);
      acc[1][t] = wmma_f16(a1, bf, acc[1][t]);
    }
  }
  // C fragment: lane<16 -> rows 0..7 (vgpr r), col=lane; lane>=16 -> rows 8..15
#pragma unroll
  for (int i = 0; i < 2; ++i) {
#pragma unroll
    for (int t = 0; t < 4; ++t) {
      size_t cbase = (size_t)(mt * 32 + i * 16 + hi * 8) * N + nt * 64 + t * 16 + ln;
#pragma unroll
      for (int r = 0; r < 8; ++r)
        C[cbase + (size_t)r * N] = (OutT)acc[i][t][r];
    }
  }
}

// ------------------------------ RoPE (in place) ----------------------------
// layout (B, S, nh, 64); thread handles one (row, head, d<32) pair.
__global__ void __launch_bounds__(256)
rope_kernel(_Float16* __restrict__ q, const float* __restrict__ cosb,
            const float* __restrict__ sinb, int nh, int total) {
  int idx = blockIdx.x * 256 + threadIdx.x;
  if (idx >= total) return;
  int d = idx & 31;
  int t = idx >> 5;
  int h = t % nh;  t /= nh;
  int s = t % SS;
  int b = t / SS;
  size_t base = (((size_t)(b * SS + s) * nh) + h) * HDD;
  float c  = cosb[s * 32 + d];
  float sn = sinb[s * 32 + d];
  float lo = (float)q[base + d];
  float hv = (float)q[base + d + 32];
  q[base + d]      = (_Float16)(lo * c - hv * sn);
  q[base + d + 32] = (_Float16)(hv * c + lo * sn);
}

// --------------------- V transpose: (B,S,KV,HD) -> (B,KV,HD,S) -------------
__global__ void __launch_bounds__(256)
vtrans_kernel(const _Float16* __restrict__ vp, _Float16* __restrict__ vt, int total) {
  int idx = blockIdx.x * 256 + threadIdx.x;
  if (idx >= total) return;
  int s = idx & (SS - 1);
  int t = idx >> 11;           // SS = 2048
  int d = t & (HDD - 1);
  t >>= 6;
  int kv = t & (KVH - 1);
  int b  = t >> 3;
  vt[idx] = vp[(((size_t)(b * SS + s) * KVH) + kv) * HDD + d];
}

// ------------------------- Flash attention (per wave) ----------------------
// Wave owns 16 query rows of TWO rep-heads that share one KV head, so every
// K-fragment and V-fragment load feeds two WMMAs. Per 32-key block:
// 8 WMMA (QK^T), online softmax x2 (cross-lane row reductions),
// P -> LDS -> A-fragment x2, 8 WMMA (P*V).
__global__ void __launch_bounds__(256)
attn_kernel(const _Float16* __restrict__ Q, const _Float16* __restrict__ Kp,
            const _Float16* __restrict__ Vt, _Float16* __restrict__ AO) {
  __shared__ __align__(32) _Float16 plds[8][2][16 * 32];
  const int lane = threadIdx.x & 31;
  const int lw   = threadIdx.x >> 5;
  const int wid  = blockIdx.x * 8 + lw;
  const int qt = wid & 127;           // S/16 = 128 query tiles
  const int pr = (wid >> 7) & 1;      // head pair within kv group
  const int kv = (wid >> 8) & 7;
  const int b  = wid >> 11;
  const int h0 = kv * 4 + pr * 2;     // REP = 4, two heads: h0, h0+1
  const int i0 = qt << 4;
  const int hi = lane >> 4;
  const int ln = lane & 15;

  // Q A-fragments (contraction halves 0..31 / 32..63) for both heads
  F16Frag aq[2][2];
#pragma unroll
  for (int p = 0; p < 2; ++p) {
    size_t qbase = (((size_t)(b * SS + i0 + ln) * HH) + h0 + p) * HDD + hi * 16;
    aq[p][0].h[0] = *(const v8h*)(Q + qbase);
    aq[p][0].h[1] = *(const v8h*)(Q + qbase + 8);
    aq[p][1].h[0] = *(const v8h*)(Q + qbase + 32);
    aq[p][1].h[1] = *(const v8h*)(Q + qbase + 40);
  }

  v8f o[2][4] = {{v8f{}, v8f{}, v8f{}, v8f{}}, {v8f{}, v8f{}, v8f{}, v8f{}}};
  float m[2][8], l[2][8];
#pragma unroll
  for (int p = 0; p < 2; ++p)
#pragma unroll
    for (int r = 0; r < 8; ++r) { m[p][r] = -1e30f; l[p][r] = 0.f; }

  const _Float16* vbase = Vt + ((size_t)b * KVH + kv) * HDD * SS;

  for (int j0 = 0; j0 < i0 + 16; j0 += 32) {
    // ---- scores: K fragments loaded once, used by both heads ----
    v8f s0[2] = {v8f{}, v8f{}};
    v8f s1[2] = {v8f{}, v8f{}};
    {
      size_t kb = (((size_t)(b * SS + j0 + ln) * KVH) + kv) * HDD + hi * 16;
      F16Frag bk0, bk1;
      bk0.h[0] = *(const v8h*)(Kp + kb);      bk0.h[1] = *(const v8h*)(Kp + kb + 8);
      bk1.h[0] = *(const v8h*)(Kp + kb + 32); bk1.h[1] = *(const v8h*)(Kp + kb + 40);
#pragma unroll
      for (int p = 0; p < 2; ++p) {
        s0[p] = wmma_f16(aq[p][0], bk0, s0[p]);
        s0[p] = wmma_f16(aq[p][1], bk1, s0[p]);
      }
    }
    {
      size_t kb = (((size_t)(b * SS + j0 + 16 + ln) * KVH) + kv) * HDD + hi * 16;
      F16Frag bk0, bk1;
      bk0.h[0] = *(const v8h*)(Kp + kb);      bk0.h[1] = *(const v8h*)(Kp + kb + 8);
      bk1.h[0] = *(const v8h*)(Kp + kb + 32); bk1.h[1] = *(const v8h*)(Kp + kb + 40);
#pragma unroll
      for (int p = 0; p < 2; ++p) {
        s1[p] = wmma_f16(aq[p][0], bk0, s1[p]);
        s1[p] = wmma_f16(aq[p][1], bk1, s1[p]);
      }
    }

    // ---- online softmax for both heads over this 32-key block ----
#pragma unroll
    for (int p = 0; p < 2; ++p) {
      _Float16* P = plds[lw][p];
      float f[8];
#pragma unroll
      for (int r = 0; r < 8; ++r) {
        int qrow = i0 + hi * 8 + r;
        float v0 = s0[p][r] * 0.125f;        // 1/sqrt(64)
        float v1 = s1[p][r] * 0.125f;
        if (j0 + ln > qrow)      v0 = -1e30f;
        if (j0 + 16 + ln > qrow) v1 = -1e30f;
        float tm = fmaxf(v0, v1);
#pragma unroll
        for (int off = 1; off < 16; off <<= 1)
          tm = fmaxf(tm, __shfl_xor(tm, off, 32));
        float nm = fmaxf(m[p][r], tm);
        float sc = __expf(m[p][r] - nm);
        m[p][r] = nm;
        float p0 = __expf(v0 - nm);
        float p1 = __expf(v1 - nm);
        float rs = p0 + p1;
#pragma unroll
        for (int off = 1; off < 16; off <<= 1)
          rs += __shfl_xor(rs, off, 32);
        l[p][r] = l[p][r] * sc + rs;
        f[r] = sc;
        // C-fragment -> LDS (row-major 16x32 f16 tile)
        P[(hi * 8 + r) * 32 + ln]      = (_Float16)p0;
        P[(hi * 8 + r) * 32 + 16 + ln] = (_Float16)p1;
      }
#pragma unroll
      for (int r = 0; r < 8; ++r) {
        o[p][0][r] *= f[r]; o[p][1][r] *= f[r];
        o[p][2][r] *= f[r]; o[p][3][r] *= f[r];
      }
    }

    // ---- P*V: V fragments loaded once, used by both heads ----
    F16Frag pa[2];
#pragma unroll
    for (int p = 0; p < 2; ++p) {
      _Float16* P = plds[lw][p];
      pa[p].h[0] = *(const v8h*)(P + ln * 32 + hi * 16);
      pa[p].h[1] = *(const v8h*)(P + ln * 32 + hi * 16 + 8);
    }
#pragma unroll
    for (int t = 0; t < 4; ++t) {
      size_t va = (size_t)(t * 16 + ln) * SS + j0 + hi * 16;
      F16Frag vb;
      vb.h[0] = *(const v8h*)(vbase + va);
      vb.h[1] = *(const v8h*)(vbase + va + 8);
      o[0][t] = wmma_f16(pa[0], vb, o[0][t]);
      o[1][t] = wmma_f16(pa[1], vb, o[1][t]);
    }
  }

  // ---- normalize and store (B,S,H,HD) f16 ----
#pragma unroll
  for (int p = 0; p < 2; ++p) {
#pragma unroll
    for (int r = 0; r < 8; ++r) {
      float inv = 1.f / l[p][r];
      size_t ob = (((size_t)(b * SS + i0 + hi * 8 + r) * HH) + h0 + p) * HDD + ln;
#pragma unroll
      for (int t = 0; t < 4; ++t)
        AO[ob + t * 16] = (_Float16)(o[p][t][r] * inv);
    }
  }
}

// ---------------------------------------------------------------------------
extern "C" void kernel_launch(void* const* d_in, const int* in_sizes, int n_in,
                              void* d_out, int out_size, void* d_ws, size_t ws_size,
                              hipStream_t stream) {
  const float* x    = (const float*)d_in[0];
  const float* cosb = (const float*)d_in[1];
  const float* sinb = (const float*)d_in[2];
  // d_in[3] = mask: unused (causal mask computed analytically)
  const float* wq = (const float*)d_in[4];
  const float* wk = (const float*)d_in[5];
  const float* wv = (const float*)d_in[6];
  const float* wo = (const float*)d_in[7];
  float* out = (float*)d_out;

  const size_t nX  = (size_t)BSs * DDim;          // 8,388,608
  const size_t nWq = (size_t)DDim * DDim;         // 4,194,304
  const size_t nWk = (size_t)(KVH * HDD) * DDim;  // 1,048,576
  const size_t nQ  = (size_t)BSs * HH * HDD;      // 8,388,608
  const size_t nK  = (size_t)BSs * KVH * HDD;     // 2,097,152

  size_t off = 0;
  auto alloc = [&](size_t elems) {
    size_t o = off;
    off = (off + elems * sizeof(_Float16) + 255) & ~(size_t)255;
    return (_Float16*)((char*)d_ws + o);
  };
  _Float16* x16  = alloc(nX);
  _Float16* wq16 = alloc(nWq);
  _Float16* wk16 = alloc(nWk);
  _Float16* wv16 = alloc(nWk);
  _Float16* wo16 = alloc(nWq);
  _Float16* Qp   = alloc(nQ);
  _Float16* Kp   = alloc(nK);
  _Float16* Vp   = alloc(nK);
  _Float16* Vt   = alloc(nK);
  _Float16* AO   = alloc(nQ);

  // 1) downconvert activations + weights to f16
  cvt_f32_f16<<<(int)(nX  / 256), 256, 0, stream>>>(x,  x16,  (int)nX);
  cvt_f32_f16<<<(int)(nWq / 256), 256, 0, stream>>>(wq, wq16, (int)nWq);
  cvt_f32_f16<<<(int)(nWk / 256), 256, 0, stream>>>(wk, wk16, (int)nWk);
  cvt_f32_f16<<<(int)(nWk / 256), 256, 0, stream>>>(wv, wv16, (int)nWk);
  cvt_f32_f16<<<(int)(nWq / 256), 256, 0, stream>>>(wo, wo16, (int)nWq);

  // 2) QKV projections (TN WMMA GEMMs, 32x64 per wave)
  gemm_tn_wmma<_Float16><<<512, 256, 0, stream>>>(x16, wq16, Qp, BSs, HH * HDD, DDim);
  gemm_tn_wmma<_Float16><<<128, 256, 0, stream>>>(x16, wk16, Kp, BSs, KVH * HDD, DDim);
  gemm_tn_wmma<_Float16><<<128, 256, 0, stream>>>(x16, wv16, Vp, BSs, KVH * HDD, DDim);

  // 3) RoPE in place on Q and K
  rope_kernel<<<(int)(BSs * HH  * 32 / 256), 256, 0, stream>>>(Qp, cosb, sinb, HH,  BSs * HH * 32);
  rope_kernel<<<(int)(BSs * KVH * 32 / 256), 256, 0, stream>>>(Kp, cosb, sinb, KVH, BSs * KVH * 32);

  // 4) V transpose -> (B, KV, HD, S)
  vtrans_kernel<<<(int)(nK / 256), 256, 0, stream>>>(Vp, Vt, (int)nK);

  // 5) flash attention: 4096 wave-jobs (2 heads each), 8 waves/block
  attn_kernel<<<512, 256, 0, stream>>>(Qp, Kp, Vt, AO);

  // 6) output projection, f32 result
  gemm_tn_wmma<float><<<512, 256, 0, stream>>>(AO, wo16, out, BSs, DDim, DDim);
}
